// ScaledDotProductAttention_69157563400295
// MI455X (gfx1250) — matrix operand order repaired and checked by reading, not compiled, not run on hardware
//
#include <hip/hip_runtime.h>

typedef __attribute__((ext_vector_type(16))) _Float16 v16h;
typedef __attribute__((ext_vector_type(8)))  _Float16 v8h;
typedef __attribute__((ext_vector_type(8)))  float    v8f;
typedef __attribute__((ext_vector_type(4)))  int      v4i;
typedef __attribute__((ext_vector_type(2)))  int      v2i;

#define DEV __device__ __forceinline__

static constexpr int   Bc    = 2, Hc = 16, Sc = 2048, Dc = 64;
static constexpr float LOG2E = 1.4426950408889634f;
static constexpr float K2    = 0.125f * LOG2E;         // (1/sqrt(64)) * log2(e), folded into Q
static constexpr float FMIN  = -3.402823466e+38f;      // finfo(f32).min

DEV v16h cat16(v8h lo, v8h hi) {
  return __builtin_shufflevector(lo, hi, 0,1,2,3,4,5,6,7,8,9,10,11,12,13,14,15);
}
DEV v8h shfl_xor16_v8h(v8h x) {
  v4i xi = __builtin_bit_cast(v4i, x), r;
#pragma unroll
  for (int j = 0; j < 4; ++j) r[j] = __shfl_xor(xi[j], 16, 32);
  return __builtin_bit_cast(v8h, r);
}
DEV v8f zero8() { v8f z = {0.f,0.f,0.f,0.f,0.f,0.f,0.f,0.f}; return z; }
DEV int pkh(float a, float b) {          // v_cvt_pk_rtz_f16_f32 -> 2 halves in a dword
  return __builtin_bit_cast(int, __builtin_amdgcn_cvt_pkrtz(a, b));
}

__global__ __launch_bounds__(256)
void fa_fwd(const float* __restrict__ Qg, const float* __restrict__ Kg,
            const float* __restrict__ Vg, const int*   __restrict__ Mg,
            float* __restrict__ Og) {
  constexpr int LDH = 72;                      // padded LDS row stride (halves)
  __shared__ __align__(16) _Float16 ldsK [64 * LDH];   // K tile,  [kv][d] f16
  __shared__ __align__(16) _Float16 ldsVT[64 * LDH];   // V tile,  [d][kv] f16 (transposed)
  __shared__ __align__(16) float    ldsBK [64];        // per-kv bias: mask? 0 : FMIN
  __shared__ __align__(16) float    ldsBALL[64];       // constant FMIN (masked-q rows)

  const int  tid  = threadIdx.x;
  const int  lane = tid & 31;
  const int  wave = tid >> 5;                  // 0..7, each owns 16 q rows
  const bool lo16 = (lane < 16);

  const int nMBlk = Sc / 128;                  // 16 query blocks per (b,h)
  const int bh    = blockIdx.x / nMBlk;
  const int mblk  = blockIdx.x % nMBlk;
  const int b     = bh / Hc;

  const float* Qb = Qg + (size_t)bh * Sc * Dc;
  const float* Kb = Kg + (size_t)bh * Sc * Dc;
  const float* Vb = Vg + (size_t)bh * Sc * Dc;
  float*       Ob = Og + (size_t)bh * Sc * Dc;
  const int*   mb = Mg + (size_t)b  * Sc;

  const int qrow = mblk * 128 + wave * 16 + (lane & 15);

  // Q^T B-fragments, pre-scaled by (1/sqrt(d))*log2(e) so WMMA emits base-2 scores.
  v16h qf[2];
#pragma unroll
  for (int c = 0; c < 2; ++c) {
    const int dbase = c * 32 + (lo16 ? 0 : 16);
    const float* qp = Qb + (size_t)qrow * Dc + dbase;
    v16h f;
#pragma unroll
    for (int j = 0; j < 4; ++j) {
      float4 x = *(const float4*)(qp + j * 4);
      f[4*j+0] = (_Float16)(x.x * K2); f[4*j+1] = (_Float16)(x.y * K2);
      f[4*j+2] = (_Float16)(x.z * K2); f[4*j+3] = (_Float16)(x.w * K2);
    }
    qf[c] = f;
  }
  const int mq = (mb[qrow] != 0);              // query-row mask (per lane-pair)
  if (tid < 64) ldsBALL[tid] = FMIN;           // constant table, written once
  // per-lane bias table base: masked-q rows read the all-FMIN table
  const float* bvbase = mq ? ldsBK : ldsBALL;

  v8f accO[4];                                 // O^T: 4 d-tiles of 16x16 f32
#pragma unroll
  for (int t = 0; t < 4; ++t) accO[t] = zero8();
  float m_run = -__builtin_inff();             // running max (base-2 domain)
  float l_run = 0.f;                           // running half-row sum

  const int g0 = lo16 ? 0 : 16, g1 = lo16 ? 8 : 24;   // A-layout K-groups

  for (int kt = 0; kt < Sc / 64; ++kt) {
    const float* Kt = Kb + (size_t)kt * 64 * Dc;
    const float* Vt = Vb + (size_t)kt * 64 * Dc;

    // ---- cooperative global -> LDS stage, packed f16 converts ----
#pragma unroll
    for (int i = 0; i < 4; ++i) {              // K: 64 rows x 16 float4
      const int id  = tid + i * 256;
      const int row = id >> 4;
      const int c4  = (id & 15) << 2;
      float4 kx = *(const float4*)(Kt + row * Dc + c4);
      v2i kk = { pkh(kx.x, kx.y), pkh(kx.z, kx.w) };
      *(v2i*)(&ldsK[row * LDH + c4]) = kk;
    }
#pragma unroll
    for (int i = 0; i < 2; ++i) {              // V: 2x4 micro-tiles, transposed store
      const int id = tid + i * 256;
      const int rp = id >> 4;                  // kv row pair 0..31
      const int c4 = (id & 15) << 2;
      float4 va = *(const float4*)(Vt + (size_t)(2*rp)     * Dc + c4);
      float4 vb = *(const float4*)(Vt + (size_t)(2*rp + 1) * Dc + c4);
      *(int*)(&ldsVT[(c4+0) * LDH + 2*rp]) = pkh(va.x, vb.x);
      *(int*)(&ldsVT[(c4+1) * LDH + 2*rp]) = pkh(va.y, vb.y);
      *(int*)(&ldsVT[(c4+2) * LDH + 2*rp]) = pkh(va.z, vb.z);
      *(int*)(&ldsVT[(c4+3) * LDH + 2*rp]) = pkh(va.w, vb.w);
    }
    const int kvb = kt * 64;
    if (tid < 64)                              // key mask -> additive bias table
      ldsBK[tid] = (mb[kvb + tid] != 0) ? 0.f : FMIN;
    if (kt + 1 < Sc / 64) {                    // prefetch next tile (global_prefetch_b8)
      __builtin_prefetch(Kt + 64 * Dc + tid * 16, 0, 3);
      __builtin_prefetch(Vt + 64 * Dc + tid * 16, 0, 3);
    }
    __syncthreads();

    // ---- S^T = K * Q^T : 4 tiles (kv rows) x 2 K-chunks of d ----
    v8f s[4];
#pragma unroll
    for (int t = 0; t < 4; ++t) {
      v8f acc = zero8();
#pragma unroll
      for (int c = 0; c < 2; ++c) {
        const _Float16* rp = &ldsK[(t * 16 + (lane & 15)) * LDH + c * 32];
        v16h a = cat16(*(const v8h*)(rp + g0), *(const v8h*)(rp + g1));
        acc = __builtin_amdgcn_wmma_f32_16x16x32_f16(false, a, false, qf[c],
                                                     (short)0, acc, false, false);
      }
      s[t] = acc;
    }

    // ---- mask via pk-add of bias table; online softmax (base-2 domain) ----
    float sv[4][8];
    float lmax = -__builtin_inff();
#pragma unroll
    for (int t = 0; t < 4; ++t) {
      const float4* bp = (const float4*)(bvbase + t * 16 + (lo16 ? 0 : 8));
      float4 b0 = bp[0], b1 = bp[1];
      const float bv[8] = {b0.x, b0.y, b0.z, b0.w, b1.x, b1.y, b1.z, b1.w};
#pragma unroll
      for (int r = 0; r < 8; ++r) {
        const float v = s[t][r] + bv[r];       // masked -> exactly FMIN
        sv[t][r] = v;
        lmax = fmaxf(lmax, v);
      }
    }
    const float tmax = fmaxf(lmax, __shfl_xor(lmax, 16, 32));
    const float mnew = fmaxf(m_run, tmax);
    const float corr = __builtin_amdgcn_exp2f(m_run - mnew);
    m_run = mnew;

    float lsum = 0.f;
    v8h ph[4];
#pragma unroll
    for (int t = 0; t < 4; ++t) {
      float pv[8];
#pragma unroll
      for (int r = 0; r < 8; ++r) {
        pv[r] = __builtin_amdgcn_exp2f(sv[t][r] - mnew);
        lsum += pv[r];
      }
      v4i pi = { pkh(pv[0], pv[1]), pkh(pv[2], pv[3]),
                 pkh(pv[4], pv[5]), pkh(pv[6], pv[7]) };
      ph[t] = __builtin_bit_cast(v8h, pi);
    }
    l_run = l_run * corr + lsum;
#pragma unroll
    for (int t = 0; t < 4; ++t) accO[t] *= corr;

    // ---- O^T += V^T * P^T  (P^T rebuilt with 4 shfl_xor16 per kv-chunk) ----
#pragma unroll
    for (int c = 0; c < 2; ++c) {
      v8h own  = lo16 ? ph[2*c]     : ph[2*c + 1];
      v8h send = lo16 ? ph[2*c + 1] : ph[2*c];
      v8h recv = shfl_xor16_v8h(send);
      v8h first  = lo16 ? own  : recv;
      v8h second = lo16 ? recv : own;
      v16h fb = cat16(first, second);          // P^T B-fragment (32 kv x 16 q)
#pragma unroll
      for (int td = 0; td < 4; ++td) {
        const _Float16* vp = &ldsVT[(td * 16 + (lane & 15)) * LDH + c * 32];
        v16h a = cat16(*(const v8h*)(vp + g0), *(const v8h*)(vp + g1));
        accO[td] = __builtin_amdgcn_wmma_f32_16x16x32_f16(false, a, false, fb,
                                                          (short)0, accO[td],
                                                          false, false);
      }
    }
    __syncthreads();
  }

  // ---- finalize: O = O_acc / l  (row sum combined across the lane pair) ----
  const float ltot = l_run + __shfl_xor(l_run, 16, 32);
  const float inv  = 1.0f / ltot;
#pragma unroll
  for (int td = 0; td < 4; ++td) {
    const int d0 = td * 16 + (lo16 ? 0 : 8);
    float* op = Ob + (size_t)qrow * Dc + d0;
    float4 o0 = {accO[td][0]*inv, accO[td][1]*inv, accO[td][2]*inv, accO[td][3]*inv};
    float4 o1 = {accO[td][4]*inv, accO[td][5]*inv, accO[td][6]*inv, accO[td][7]*inv};
    *(float4*)(op)     = o0;
    *(float4*)(op + 4) = o1;
  }
}

extern "C" void kernel_launch(void* const* d_in, const int* in_sizes, int n_in,
                              void* d_out, int out_size, void* d_ws, size_t ws_size,
                              hipStream_t stream) {
  const float* Q = (const float*)d_in[0];
  const float* K = (const float*)d_in[1];
  const float* V = (const float*)d_in[2];
  const int*   M = (const int*)  d_in[3];
  float* O = (float*)d_out;
  (void)in_sizes; (void)n_in; (void)out_size; (void)d_ws; (void)ws_size;

  const int grid = Bc * Hc * (Sc / 128);       // 512 workgroups, 8 waves each
  fa_fwd<<<grid, 256, 0, stream>>>(Q, K, V, M, O);
}
